// FP_Upsampler_27943057228024
// MI455X (gfx1250) — matrix operand order repaired and checked by reading, not compiled, not run on hardware
//
#include <hip/hip_runtime.h>
#include <hip/hip_bf16.h>
#include <math.h>

typedef __attribute__((ext_vector_type(16))) _Float16 v16h;
typedef __attribute__((ext_vector_type(8)))  float    v8f;

#define NQ     40000
#define NS     20000
#define CDIM   64
#define HDIM   128
#define TILE   1024
#define LN_EPS 1e-5f
#define EPSW   1e-8f

// ---------------------------------------------------------------------------
// Kernel 1: mean / std (ddof=1) of concat(gt_coords, coords) per dimension.
// Single block, deterministic tree reduction. stats[0..2]=mu, stats[3..5]=sd.
// ---------------------------------------------------------------------------
__global__ void stats_kernel(const float* __restrict__ coords,
                             const float* __restrict__ gt,
                             float* __restrict__ stats) {
    __shared__ float red[1024][6];
    float s[6] = {0.f, 0.f, 0.f, 0.f, 0.f, 0.f};
    for (int i = threadIdx.x; i < NQ + NS; i += 1024) {
        const float* p = (i < NQ) ? (gt + i * 3) : (coords + (i - NQ) * 3);
        float x = p[0], y = p[1], z = p[2];
        s[0] += x; s[1] += y; s[2] += z;
        s[3] += x * x; s[4] += y * y; s[5] += z * z;
    }
    #pragma unroll
    for (int d = 0; d < 6; d++) red[threadIdx.x][d] = s[d];
    __syncthreads();
    for (int stride = 512; stride > 0; stride >>= 1) {
        if (threadIdx.x < (unsigned)stride) {
            #pragma unroll
            for (int d = 0; d < 6; d++)
                red[threadIdx.x][d] += red[threadIdx.x + stride][d];
        }
        __syncthreads();
    }
    if (threadIdx.x == 0) {
        const float n = (float)(NQ + NS);
        #pragma unroll
        for (int d = 0; d < 3; d++) {
            float mu  = red[0][d] / n;
            float var = (red[0][3 + d] - n * mu * mu) / (n - 1.f); // ddof=1
            float sd  = sqrtf(fmaxf(var, 0.f));
            if (fabsf(sd) < EPSW) sd = 1.f;
            stats[d]     = mu;
            stats[3 + d] = sd;
        }
    }
}

// ---------------------------------------------------------------------------
// Kernel 2: pre-swizzle W1/W2 into f16 *fragment-major* WMMA B layout, and
// copy gt_coords into d_out[0 : 120000].
//
// Fragment-major layout: frag id f, lane l, half h stored at f*512 + l*16 + h.
//   GEMM1 (W1, 64x128): f = nt*2 + kc  (nt<8, kc<2)
//       value = W1[(kc*32 + (l>>4)*16 + h) * 128 + nt*16 + (l&15)]
//   GEMM2 (W2, 128x64): f = nt*4 + kc  (nt<4, kc<4)
//       value = W2[(kc*32 + (l>>4)*16 + h) * 64  + nt*16 + (l&15)]
// This lets every lane fetch its whole B fragment as one 32-byte load.
// ---------------------------------------------------------------------------
__global__ void prep_kernel(const float* __restrict__ W1,
                            const float* __restrict__ W2,
                            const float* __restrict__ gt,
                            _Float16* __restrict__ w1f,
                            _Float16* __restrict__ w2f,
                            float* __restrict__ out_coords) {
    int i = blockIdx.x * blockDim.x + threadIdx.x;
    if (i < CDIM * HDIM) {
        const int f    = i >> 9;          // fragment id (16 per weight)
        const int r    = i & 511;
        const int lane = r >> 4;
        const int h    = r & 15;
        const int nlo  = lane & 15;
        const int lg   = lane >> 4;
        {   // W1: f = nt*2 + kc
            const int nt = f >> 1, kc = f & 1;
            const int k  = kc * 32 + lg * 16 + h;
            const int n  = nt * 16 + nlo;
            w1f[i] = (_Float16)W1[k * HDIM + n];
        }
        {   // W2: f = nt*4 + kc
            const int nt = f >> 2, kc = f & 3;
            const int k  = kc * 32 + lg * 16 + h;
            const int n  = nt * 16 + nlo;
            w2f[i] = (_Float16)W2[k * CDIM + n];
        }
    }
    if (i < NQ * 3) out_coords[i] = gt[i];
}

// ---------------------------------------------------------------------------
// Kernel 3: brute-force KNN (K=3) + softmax-weighted feature interpolation.
// LDS tiles of 1024 standardized source points; per-thread sorted top-3.
// ---------------------------------------------------------------------------
__global__ void knn_kernel(const float* __restrict__ coords,
                           const float* __restrict__ feats,
                           const float* __restrict__ gt,
                           const float* __restrict__ stats,
                           float* __restrict__ fi) {
    __shared__ float sx[TILE], sy[TILE], sz[TILE];
    const int m = blockIdx.x * blockDim.x + threadIdx.x;
    const bool act = (m < NQ);

    const float mu0 = stats[0], mu1 = stats[1], mu2 = stats[2];
    const float r0s = 1.f / stats[3], r1s = 1.f / stats[4], r2s = 1.f / stats[5];

    float qx = 0.f, qy = 0.f, qz = 0.f;
    if (act) {
        qx = (gt[m * 3 + 0] - mu0) * r0s;
        qy = (gt[m * 3 + 1] - mu1) * r1s;
        qz = (gt[m * 3 + 2] - mu2) * r2s;
    }

    float b0 = 3.4e38f, b1v = 3.4e38f, b2v = 3.4e38f;
    int   j0 = 0, j1 = 0, j2 = 0;

    for (int t0 = 0; t0 < NS; t0 += TILE) {
        const int cnt = min(TILE, NS - t0);
        for (int j = threadIdx.x; j < cnt; j += blockDim.x) {
            const float* p = coords + (t0 + j) * 3;
            sx[j] = (p[0] - mu0) * r0s;
            sy[j] = (p[1] - mu1) * r1s;
            sz[j] = (p[2] - mu2) * r2s;
        }
        __syncthreads();
        if (act) {
            #pragma unroll 4
            for (int j = 0; j < cnt; j++) {
                float dx = qx - sx[j], dy = qy - sy[j], dz = qz - sz[j];
                float dd = fmaf(dx, dx, fmaf(dy, dy, dz * dz));
                int   id = t0 + j;
                if (dd < b2v) {
                    if (dd < b1v) {
                        b2v = b1v; j2 = j1;
                        if (dd < b0) { b1v = b0; j1 = j0; b0 = dd; j0 = id; }
                        else         { b1v = dd; j1 = id; }
                    } else { b2v = dd; j2 = id; }
                }
            }
        }
        __syncthreads();
    }

    if (act) {
        float d0 = sqrtf(fmaxf(b0, 0.f));
        float d1 = sqrtf(fmaxf(b1v, 0.f));
        float d2 = sqrtf(fmaxf(b2v, 0.f));
        float scale = fmaxf((d0 + d1 + d2) * (1.f / 3.f), EPSW);
        float inv_s = 1.f / scale;
        float w0 = 1.f;                          // exp(-(d0-d0)/scale)
        float w1 = expf(-(d1 - d0) * inv_s);
        float w2 = expf(-(d2 - d0) * inv_s);
        float invw = 1.f / (w0 + w1 + w2);
        w0 *= invw; w1 *= invw; w2 *= invw;

        const float4* f0 = (const float4*)(feats + (size_t)j0 * CDIM);
        const float4* f1 = (const float4*)(feats + (size_t)j1 * CDIM);
        const float4* f2 = (const float4*)(feats + (size_t)j2 * CDIM);
        float4* o = (float4*)(fi + (size_t)m * CDIM);
        #pragma unroll
        for (int c = 0; c < CDIM / 4; c++) {
            float4 a = f0[c], b = f1[c], cc = f2[c], r;
            r.x = w0 * a.x + w1 * b.x + w2 * cc.x;
            r.y = w0 * a.y + w1 * b.y + w2 * cc.y;
            r.z = w0 * a.z + w1 * b.z + w2 * cc.z;
            r.w = w0 * a.w + w1 * b.w + w2 * cc.w;
            o[c] = r;
        }
    }
}

// ---------------------------------------------------------------------------
// Helper: convert two float4 runs into 8 consecutive halves of a v16h frag.
// ---------------------------------------------------------------------------
__device__ __forceinline__ void put8(v16h& v, int base, float4 a, float4 b) {
    v[base + 0] = (_Float16)a.x; v[base + 1] = (_Float16)a.y;
    v[base + 2] = (_Float16)a.z; v[base + 3] = (_Float16)a.w;
    v[base + 4] = (_Float16)b.x; v[base + 5] = (_Float16)b.y;
    v[base + 6] = (_Float16)b.z; v[base + 7] = (_Float16)b.w;
}

// ---------------------------------------------------------------------------
// Kernel 4: fused MLP via v_wmma_f32_16x16x32_f16.
// One wave per 16-row tile: GEMM1 (16x64 @ 64x128) -> bias+ReLU -> LN ->
// GEMM2 (16x128 @ 128x64) -> bias + residual(fi) -> LN -> tanh*6 -> d_out.
// B fragments are pre-swizzled (one v16h load each); A fed by float4 loads.
// ---------------------------------------------------------------------------
__global__ void mlp_kernel(const float* __restrict__ fi,
                           const _Float16* __restrict__ w1f,
                           const _Float16* __restrict__ w2f,
                           const float* __restrict__ b1,
                           const float* __restrict__ g1,
                           const float* __restrict__ be1,
                           const float* __restrict__ b2,
                           const float* __restrict__ gf,
                           const float* __restrict__ bf,
                           float* __restrict__ out) {
    __shared__ float hbuf[4][16 * HDIM];          // 32 KB, one region per wave
    const int lane = threadIdx.x & 31;
    const int wave = threadIdx.x >> 5;
    const int tile = blockIdx.x * 4 + wave;
    const bool act = (tile < NQ / 16);            // wave-uniform: EXEC all-1s
    const int m0   = tile * 16;
    const int nlo  = lane & 15;                   // column / row-in-tile index
    const int lg   = lane >> 4;                   // lane-group (K/M half select)
    float* H = hbuf[wave];

    // ---- GEMM1: fi[16x64] @ W1[64x128] ----
    if (act) {
        // A fragments (16x32 f16): half h -> k = kc*32 + lg*16 + h
        // Runs of 8 consecutive K values -> float4 pairs.
        const float4* arow4 = (const float4*)(fi + (size_t)(m0 + nlo) * CDIM);
        v16h a0, a1;
        {
            const int q = lg * 4;                 // (lg*16)/4
            put8(a0, 0, arow4[q + 0],  arow4[q + 1]);   // kc0, g0
            put8(a0, 8, arow4[q + 2],  arow4[q + 3]);   // kc0, g1
            put8(a1, 0, arow4[q + 8],  arow4[q + 9]);   // kc1, g0
            put8(a1, 8, arow4[q + 10], arow4[q + 11]);  // kc1, g1
        }
        const v16h* w1v = (const v16h*)(w1f + (size_t)lane * 16);  // + f*512
        #pragma unroll
        for (int nt = 0; nt < HDIM / 16; nt++) {
            const int n = nt * 16 + nlo;
            v16h bf0 = w1v[(nt * 2 + 0) * 32];    // stride 512 halves = 32 v16h
            v16h bf1 = w1v[(nt * 2 + 1) * 32];
            v8f acc = {};
            acc = __builtin_amdgcn_wmma_f32_16x16x32_f16(false, a0, false, bf0,
                                                         (short)0, acc, false, false);
            acc = __builtin_amdgcn_wmma_f32_16x16x32_f16(false, a1, false, bf1,
                                                         (short)0, acc, false, false);
            const float bias = b1[n];
            #pragma unroll
            for (int r = 0; r < 8; r++)           // C/D: row = r + 8*lg, col = n
                H[(r + 8 * lg) * HDIM + n] = fmaxf(acc[r] + bias, 0.f);
        }
    }
    __syncthreads();

    // ---- LayerNorm over HDIM (lanes 0..15 each own one row) ----
    if (act && lane < 16) {
        float4* row4 = (float4*)(H + lane * HDIM);
        float s = 0.f, ss = 0.f;
        #pragma unroll
        for (int j = 0; j < HDIM / 4; j++) {
            float4 v = row4[j];
            s  += v.x + v.y + v.z + v.w;
            ss += v.x * v.x + v.y * v.y + v.z * v.z + v.w * v.w;
        }
        float mu  = s * (1.f / HDIM);
        float var = ss * (1.f / HDIM) - mu * mu;
        float inv = rsqrtf(fmaxf(var, 0.f) + LN_EPS);
        const float4* g4 = (const float4*)g1;
        const float4* e4 = (const float4*)be1;
        #pragma unroll
        for (int j = 0; j < HDIM / 4; j++) {
            float4 v = row4[j], g = g4[j], e = e4[j], r;
            r.x = (v.x - mu) * inv * g.x + e.x;
            r.y = (v.y - mu) * inv * g.y + e.y;
            r.z = (v.z - mu) * inv * g.z + e.z;
            r.w = (v.w - mu) * inv * g.w + e.w;
            row4[j] = r;
        }
    }
    __syncthreads();

    // ---- A2 fragments for GEMM2 (16x128, 4 K-steps) from LDS ----
    v16h a2[4];
    if (act) {
        const float4* arow4 = (const float4*)(H + nlo * HDIM);
        const int q = lg * 4;
        #pragma unroll
        for (int kc = 0; kc < 4; kc++) {
            put8(a2[kc], 0, arow4[kc * 8 + q + 0], arow4[kc * 8 + q + 1]);
            put8(a2[kc], 8, arow4[kc * 8 + q + 2], arow4[kc * 8 + q + 3]);
        }
    }
    __syncthreads();                              // H free: reuse as FO (16x64)

    // ---- GEMM2: h[16x128] @ W2[128x64], + b2 + residual fi ----
    float* FO = H;
    if (act) {
        const v16h* w2v = (const v16h*)(w2f + (size_t)lane * 16);
        #pragma unroll
        for (int nt = 0; nt < CDIM / 16; nt++) {
            const int n = nt * 16 + nlo;
            v8f acc = {};
            #pragma unroll
            for (int kc = 0; kc < 4; kc++) {
                v16h bfr = w2v[(nt * 4 + kc) * 32];
                acc = __builtin_amdgcn_wmma_f32_16x16x32_f16(false, a2[kc], false, bfr,
                                                             (short)0, acc, false, false);
            }
            const float bias = b2[n];
            #pragma unroll
            for (int r = 0; r < 8; r++) {
                int row = r + 8 * lg;
                FO[row * CDIM + n] =
                    acc[r] + bias + fi[(size_t)(m0 + row) * CDIM + n];
            }
        }
    }
    __syncthreads();

    // ---- LayerNorm over CDIM + tanh*CLAMP + store ----
    if (act && lane < 16) {
        float4* row4 = (float4*)(FO + lane * CDIM);
        float s = 0.f, ss = 0.f;
        #pragma unroll
        for (int j = 0; j < CDIM / 4; j++) {
            float4 v = row4[j];
            s  += v.x + v.y + v.z + v.w;
            ss += v.x * v.x + v.y * v.y + v.z * v.z + v.w * v.w;
        }
        float mu  = s * (1.f / CDIM);
        float var = ss * (1.f / CDIM) - mu * mu;
        float inv = rsqrtf(fmaxf(var, 0.f) + LN_EPS);
        const float4* g4 = (const float4*)gf;
        const float4* e4 = (const float4*)bf;
        float4* o4 = (float4*)(out + (size_t)(m0 + lane) * CDIM);
        #pragma unroll
        for (int j = 0; j < CDIM / 4; j++) {
            float4 v = row4[j], g = g4[j], e = e4[j], r;
            r.x = tanhf((v.x - mu) * inv * g.x + e.x) * 6.f;
            r.y = tanhf((v.y - mu) * inv * g.y + e.y) * 6.f;
            r.z = tanhf((v.z - mu) * inv * g.z + e.z) * 6.f;
            r.w = tanhf((v.w - mu) * inv * g.w + e.w) * 6.f;
            o4[j] = r;
        }
    }
}

// ---------------------------------------------------------------------------
extern "C" void kernel_launch(void* const* d_in, const int* in_sizes, int n_in,
                              void* d_out, int out_size, void* d_ws, size_t ws_size,
                              hipStream_t stream) {
    const float* coords = (const float*)d_in[0];   // [20000,3]
    const float* feats  = (const float*)d_in[1];   // [20000,64]
    const float* gt     = (const float*)d_in[2];   // [40000,3]
    const float* W1     = (const float*)d_in[3];   // [64,128]
    const float* b1     = (const float*)d_in[4];
    const float* g1     = (const float*)d_in[5];
    const float* be1    = (const float*)d_in[6];
    const float* W2     = (const float*)d_in[7];   // [128,64]
    const float* b2     = (const float*)d_in[8];
    const float* gf     = (const float*)d_in[9];
    const float* bf     = (const float*)d_in[10];

    float* out = (float*)d_out;                    // [40000*3] ++ [40000*64]

    char* ws = (char*)d_ws;
    float*    stats = (float*)ws;                             // 64 B
    _Float16* w1f   = (_Float16*)(ws + 64);                   // 16 KB frag-major
    _Float16* w2f   = (_Float16*)(ws + 64 + 16384);           // 16 KB frag-major
    float*    fi    = (float*)(ws + 64 + 32768);              // 40000*64 f32

    stats_kernel<<<1, 1024, 0, stream>>>(coords, gt, stats);
    prep_kernel<<<(NQ * 3 + 255) / 256, 256, 0, stream>>>(W1, W2, gt, w1f, w2f, out);
    knn_kernel<<<(NQ + 255) / 256, 256, 0, stream>>>(coords, feats, gt, stats, fi);
    mlp_kernel<<<(NQ / 16 + 3) / 4, 128, 0, stream>>>(fi, w1f, w2f,
                                                      b1, g1, be1, b2, gf, bf,
                                                      out + NQ * 3);
}
